// MultiImgCAModel_80891414053321
// MI455X (gfx1250) — matrix operand order repaired and checked by reading, not compiled, not run on hardware
//
#include <hip/hip_runtime.h>
#include <hip/hip_bf16.h>
#include <stdint.h>

typedef _Float16 f16;
typedef __attribute__((ext_vector_type(16))) _Float16 v16h;
typedef __attribute__((ext_vector_type(8)))  float    v8f;
typedef __attribute__((ext_vector_type(4)))  unsigned int u32x4;
typedef __attribute__((ext_vector_type(8)))  int          i32x8;
typedef __attribute__((ext_vector_type(4)))  int          i32x4;

#define BN_EPS 1e-3f
#define NB   128
#define CI   4
#define HIDC 12
#define IMGC 72
#define POSTC 76
#define K1   228       // POSTC*3
#define FLAT 10368     // 18*18*32
#define ND   864       // HIDC*IMGC

// ---- workspace byte offsets (total ~76.1 MB) ----
#define WS_H1     0u          // 128*36*36*32 f32   = 21,233,664 B
#define WS_XNEW   21233664u   // 128*64*64*16 f32   = 33,554,432 B
#define WS_FEATA  54788096u   // 128*10368 f16      =  2,654,208 B
#define WS_FILTBT 57442304u   // 128*80*32 f16      =    655,360 B
#define WS_W1T    58097664u   // 128*256 f16        =     65,536 B
#define WS_W2T    58163200u   // 16*128 f16         =      4,096 B  (contiguous after W1T)
#define WS_DWT    58167296u   // 864*10368 f16      = 17,915,904 B

union Frag {
    v16h v;
    f16  e[16];
    uint4 q[2];
};

// 16-bit A-matrix 16x32 layout: element e of lane (half = lane/16) -> K index.
__device__ __forceinline__ constexpr int kOfElemA(int e, int half) {
    int j = e >> 1, w = e & 1;
    return (j < 4) ? (8 * half + 2 * j + w) : (16 + 8 * half + 2 * (j - 4) + w);
}

__device__ __forceinline__ float fireHash(uint32_t i) {
    i ^= i >> 17; i *= 0xed5ad4bbu;
    i ^= i >> 11; i *= 0xac4c1b51u;
    i ^= i >> 15; i *= 0x31848babu;
    i ^= i >> 14;
    return (float)(i >> 8) * (1.0f / 16777216.0f);
}

// ---- Tensor Data Mover: 1-row copy of n_dw dwords global -> LDS ------------
// D# group0: count=1 | lds_addr | global_addr[56:0] | type=2
// D# group1: data_size=4B, tensor_dim0=tile_dim0=n_dw, tensor_dim1=tile_dim1=1,
//            tensor_dim0_stride=n_dw
__device__ __forceinline__ void tdm_copy_to_lds(uint32_t lds_addr,
                                                const void* gptr, uint32_t n_dw) {
    uint64_t ga = (uint64_t)(uintptr_t)gptr;
    u32x4 g0 = { 1u,                                   // count=1, user mode
                 lds_addr,
                 (unsigned)(ga & 0xffffffffu),
                 (unsigned)((ga >> 32) & 0x01ffffffu) | 0x80000000u }; // type=2
    i32x8 g1 = { (int)0x20000u,                        // data_size=2 (4 bytes)
                 (int)((n_dw & 0xffffu) << 16),        // tensor_dim0[15:0]
                 (int)(((n_dw >> 16) & 0xffffu) | 0x10000u), // dim0[31:16], dim1=1
                 (int)((n_dw & 0xffffu) << 16),        // tile_dim0
                 1,                                    // tile_dim1=1
                 (int)n_dw,                            // tensor_dim0_stride[31:0]
                 0, 0 };
    i32x4 gz = { 0, 0, 0, 0 };
#if defined(__clang_major__) && (__clang_major__ >= 23)
    i32x8 gz8 = { 0, 0, 0, 0, 0, 0, 0, 0 };
    __builtin_amdgcn_tensor_load_to_lds(g0, g1, gz, gz, gz8, 0);
#else
    __builtin_amdgcn_tensor_load_to_lds(g0, g1, gz, gz, 0);
#endif
}

// ---------------- k_init: transposed/padded f16 weights + zero filter pad ----
__global__ void k_init(const float* __restrict__ w1, const float* __restrict__ w2,
                       f16* __restrict__ w1t, f16* __restrict__ w2t,
                       f16* __restrict__ filtbt) {
    int i = blockIdx.x * 256 + threadIdx.x;
    if (i < 128 * 256) {           // w1t[d][k], k padded 228->256
        int d = i >> 8, k = i & 255;
        w1t[i] = (k < K1) ? (f16)w1[k * 128 + d] : (f16)0.f;
    }
    if (i < 16 * 128) {            // w2t[n][k]
        int n = i >> 7, k = i & 127;
        w2t[i] = (f16)w2[k * 16 + n];
    }
    if (i < 128 * 80 * 32) filtbt[i] = (f16)0.f;
}

// ---------------- k_wtrans: dense_w (10368x864 f32) -> dwT[n][k] f16 --------
__global__ void k_wtrans(const float* __restrict__ dw, f16* __restrict__ dwt) {
    int i = blockIdx.x * 256 + threadIdx.x;
    if (i >= ND * FLAT) return;
    int n = i / FLAT, k = i % FLAT;
    dwt[i] = (f16)dw[k * ND + n];
}

// ---------------- k_conv1: bn1 -> conv3x3(4->32) -> relu -> maxpool2 --------
__global__ void k_conv1(const float* __restrict__ img,
                        const float* __restrict__ g1, const float* __restrict__ b1,
                        const float* __restrict__ m1, const float* __restrict__ v1,
                        const float* __restrict__ w, const float* __restrict__ bias,
                        float* __restrict__ h1) {
    int idx = blockIdx.x * 128 + threadIdx.x;
    if (idx >= NB * 36 * 36 * 32) return;
    int oc = idx & 31; int t = idx >> 5;
    int ow = t % 36; t /= 36; int oh = t % 36; int b = t / 36;
    float sc[CI], sh[CI];
    #pragma unroll
    for (int c = 0; c < CI; c++) {
        float s = rsqrtf(v1[c] + BN_EPS) * g1[c];
        sc[c] = s; sh[c] = b1[c] - m1[c] * s;
    }
    float best = -1e30f;
    for (int ph = 0; ph < 2; ++ph)
    for (int pw = 0; pw < 2; ++pw) {
        int ih = 2 * oh + ph, iw = 2 * ow + pw;
        float acc = bias[oc];
        for (int dy = 0; dy < 3; ++dy) {
            int y = ih + dy - 1; if ((unsigned)y >= 72u) continue;
            for (int dx = 0; dx < 3; ++dx) {
                int x = iw + dx - 1; if ((unsigned)x >= 72u) continue;
                float4 iv = *(const float4*)(img + (((b * 72) + y) * 72 + x) * CI);
                const float* wp = w + ((dy * 3 + dx) * CI) * 32 + oc;
                acc += (iv.x * sc[0] + sh[0]) * wp[0];
                acc += (iv.y * sc[1] + sh[1]) * wp[32];
                acc += (iv.z * sc[2] + sh[2]) * wp[64];
                acc += (iv.w * sc[3] + sh[3]) * wp[96];
            }
        }
        best = fmaxf(best, fmaxf(acc, 0.f));
    }
    h1[idx] = best;
}

// ---------------- k_conv2: bn2 -> conv3x3(32->32) -> relu -> pool -> f16 ----
__global__ void k_conv2(const float* __restrict__ h1,
                        const float* __restrict__ g, const float* __restrict__ be,
                        const float* __restrict__ m, const float* __restrict__ v,
                        const float* __restrict__ w, const float* __restrict__ bias,
                        f16* __restrict__ featA) {
    int idx = blockIdx.x * 128 + threadIdx.x;
    if (idx >= NB * 18 * 18 * 32) return;
    int oc = idx & 31; int t = idx >> 5;
    int ow = t % 18; t /= 18; int oh = t % 18; int b = t / 18;
    float best = -1e30f;
    for (int ph = 0; ph < 2; ++ph)
    for (int pw = 0; pw < 2; ++pw) {
        int ih = 2 * oh + ph, iw = 2 * ow + pw;
        float acc = bias[oc];
        for (int c = 0; c < 32; c++) {
            float s = rsqrtf(v[c] + BN_EPS) * g[c];
            float sh = be[c] - m[c] * s;
            const float* wp = w + c * 32 + oc;
            for (int dy = 0; dy < 3; ++dy) {
                int y = ih + dy - 1; if ((unsigned)y >= 36u) continue;
                for (int dx = 0; dx < 3; ++dx) {
                    int x = iw + dx - 1; if ((unsigned)x >= 36u) continue;
                    acc += (h1[(((b * 36) + y) * 36 + x) * 32 + c] * s + sh)
                           * wp[(dy * 3 + dx) * 1024];
                }
            }
        }
        best = fmaxf(best, fmaxf(acc, 0.f));
    }
    featA[b * FLAT + (oh * 18 + ow) * 32 + oc] = (f16)best;
}

// ---------------- k_dense: WMMA GEMM 128x10368x864, relu, scatter filters ---
__global__ void __launch_bounds__(128) k_dense(const f16* __restrict__ featA,
                                               const f16* __restrict__ dwt,
                                               const float* __restrict__ db,
                                               f16* __restrict__ filtbt) {
    int lane = threadIdx.x & 31, wib = threadIdx.x >> 5;
    int wid = blockIdx.x * 4 + wib;     // 0..431 = 8 M-tiles x 54 N-tiles
    int mt = wid / 54, nt = wid % 54;
    int half = lane >> 4, lm = lane & 15;
    int n = nt * 16 + lm;
    const f16* arow = featA + (mt * 16 + lm) * FLAT;
    const f16* brow = dwt + n * FLAT;
    v8f c = {0.f, 0.f, 0.f, 0.f, 0.f, 0.f, 0.f, 0.f};
    for (int s = 0; s < FLAT / 32; s++) {
        int ks = s * 32;
        Frag a, bf;
        a.q[0] = *(const uint4*)(arow + ks + 8 * half);
        a.q[1] = *(const uint4*)(arow + ks + 16 + 8 * half);
        bf.q[0] = *(const uint4*)(brow + ks + 8 * half);
        bf.q[1] = *(const uint4*)(brow + ks + 16 + 8 * half);
        c = __builtin_amdgcn_wmma_f32_16x16x32_f16(false, a.v, false, bf.v,
                                                   (short)0, c, false, false);
    }
    float bias = db[n];
    int cidx = n / IMGC, o = n % IMGC;
    #pragma unroll
    for (int r = 0; r < 8; r++) {
        float vv = fmaxf(c[r] + bias, 0.f);
        int batch = mt * 16 + r + 8 * half;
        filtbt[batch * 2560 + o * 32 + cidx] = (f16)vv;   // [b][n=80 pad][k=32 pad]
    }
}

// ---------------- k_update: img_conv(WMMA) + perception + MLP(WMMA) ---------
__global__ void __launch_bounds__(128) k_update(const float* __restrict__ x,
                                                const f16* __restrict__ filtbt,
                                                const f16* __restrict__ w1t,
                                                const float* __restrict__ ub1,
                                                const float* __restrict__ ub2,
                                                float* __restrict__ xnew) {
    __shared__ f16 sFilt[2560];          // [n=80][k=32]
    __shared__ f16 sHalo[3 * 66 * 80];   // [row][1+64+1 cols][80 ch pad]
    __shared__ f16 sPerc[4 * 16 * 256];  // [tile][px][K1 pad 256]
    __shared__ f16 sU[4 * 16 * 128];     // [tile][px][128]
    __shared__ f16 sW12[32768 + 2048];   // w1t (128x256) + w2t (16x128)

    int tid = threadIdx.x;
    int lane = tid & 31, wib = tid >> 5;
    int half = lane >> 4, lm = lane & 15;
    int rowId = blockIdx.x;              // one board row per block
    int b = rowId >> 6, h = rowId & 63;

    __builtin_prefetch(x + ((b * 64 + h) * 64) * 16, 0, 1);

    // TDM: async DMA of operand blocks into LDS (wave 0 issues, all wait at barrier)
    if (wib == 0) {
        tdm_copy_to_lds((uint32_t)(uintptr_t)sFilt, filtbt + b * 2560, 2560 / 2);
        tdm_copy_to_lds((uint32_t)(uintptr_t)sW12, w1t, (32768 + 2048) / 2);
        __builtin_amdgcn_s_wait_tensorcnt(0);
    }
    // zero halo + perception pad; redundant VALU copy of same operand bytes
    // (keeps semantics if the TDM descriptor path is inert)
    {
        uint32_t* z1 = (uint32_t*)sHalo;
        for (int i = tid; i < 3 * 66 * 80 / 2; i += 128) z1[i] = 0u;
        uint32_t* z2 = (uint32_t*)sPerc;
        for (int i = tid; i < 4 * 16 * 256 / 2; i += 128) z2[i] = 0u;
        const uint32_t* fs = (const uint32_t*)(filtbt + b * 2560);
        uint32_t* fd = (uint32_t*)sFilt;
        for (int i = tid; i < 1280; i += 128) fd[i] = fs[i];
        const uint32_t* ws_ = (const uint32_t*)w1t;   // w2t is contiguous after w1t
        uint32_t* wd = (uint32_t*)sW12;
        for (int i = tid; i < (32768 + 2048) / 2; i += 128) wd[i] = ws_[i];
    }
    __syncthreads();

    // phase 1: img_aware halo rows h-1..h+1 via WMMA (K=12 pad 32, N=72 pad 80)
    for (int tIdx = wib; tIdx < 12; tIdx += 4) {
        int hr = tIdx >> 2;              // halo row 0..2
        int hh = h - 1 + hr;
        int w0 = (tIdx & 3) * 16;
        if ((unsigned)hh < 64u) {
            union { float4 q[3]; float s[12]; } hv;
            const float* hp = x + ((b * 64 + hh) * 64 + (w0 + lm)) * 16 + 4;
            hv.q[0] = *(const float4*)(hp);
            hv.q[1] = *(const float4*)(hp + 4);
            hv.q[2] = *(const float4*)(hp + 8);
            Frag a;
            #pragma unroll
            for (int e = 0; e < 16; e++) {
                constexpr int dummy = 0; (void)dummy;
                int k0 = kOfElemA(e, 0), k1 = kOfElemA(e, 1);
                float v0 = (k0 < HIDC) ? hv.s[k0] : 0.f;
                float v1 = (k1 < HIDC) ? hv.s[k1] : 0.f;
                a.e[e] = (f16)(half ? v1 : v0);
            }
            #pragma unroll
            for (int t = 0; t < 5; t++) {
                Frag bf;
                int n = t * 16 + lm;
                bf.q[0] = *(const uint4*)(sFilt + n * 32 + 8 * half);
                bf.q[1] = *(const uint4*)(sFilt + n * 32 + 16 + 8 * half);
                v8f c = {0.f, 0.f, 0.f, 0.f, 0.f, 0.f, 0.f, 0.f};
                c = __builtin_amdgcn_wmma_f32_16x16x32_f16(false, a.v, false, bf.v,
                                                           (short)0, c, false, false);
                if (n < IMGC) {
                    #pragma unroll
                    for (int r = 0; r < 8; r++) {
                        int p2 = r + 8 * half;
                        sHalo[(hr * 66 + 1 + w0 + p2) * 80 + 4 + n] = (f16)c[r];
                    }
                }
            }
        }
    }
    // board channels 0..3 into halo
    for (int i = tid; i < 3 * 64; i += 128) {
        int hr = i >> 6, px = i & 63;
        int hh = h - 1 + hr;
        if ((unsigned)hh < 64u) {
            float4 bc = *(const float4*)(x + ((b * 64 + hh) * 64 + px) * 16);
            union { f16 e[4]; uint2 u; } pk;
            pk.e[0] = (f16)bc.x; pk.e[1] = (f16)bc.y;
            pk.e[2] = (f16)bc.z; pk.e[3] = (f16)bc.w;
            *(uint2*)(sHalo + (hr * 66 + 1 + px) * 80) = pk.u;
        }
    }
    __syncthreads();

    // phase 2: depthwise perception (ident / sobel-x / sobel-y) -> A matrix LDS
    for (int i = tid; i < 64 * POSTC; i += 128) {
        int px = i / POSTC, g = i % POSTC;
        float a00 = (float)sHalo[(0 * 66 + px    ) * 80 + g];
        float a01 = (float)sHalo[(0 * 66 + px + 1) * 80 + g];
        float a02 = (float)sHalo[(0 * 66 + px + 2) * 80 + g];
        float a10 = (float)sHalo[(1 * 66 + px    ) * 80 + g];
        float a11 = (float)sHalo[(1 * 66 + px + 1) * 80 + g];
        float a12 = (float)sHalo[(1 * 66 + px + 2) * 80 + g];
        float a20 = (float)sHalo[(2 * 66 + px    ) * 80 + g];
        float a21 = (float)sHalo[(2 * 66 + px + 1) * 80 + g];
        float a22 = (float)sHalo[(2 * 66 + px + 2) * 80 + g];
        float sx = (-a00 + a02 - 2.f * a10 + 2.f * a12 - a20 + a22) * 0.125f;
        float sy = (-a00 - 2.f * a01 - a02 + a20 + 2.f * a21 + a22) * 0.125f;
        f16* pp = sPerc + ((px >> 4) * 16 + (px & 15)) * 256 + 3 * g;
        pp[0] = (f16)a11; pp[1] = (f16)sx; pp[2] = (f16)sy;
    }
    __syncthreads();

    // phase 3: per-wave 16-pixel tile MLP 228->128->16 (all operands in LDS)
    {
        int tw = wib;
        const f16* percBase = sPerc + tw * 16 * 256;
        v8f zero = {0.f, 0.f, 0.f, 0.f, 0.f, 0.f, 0.f, 0.f};
        v8f acc[8];
        #pragma unroll
        for (int t = 0; t < 8; t++) acc[t] = zero;
        for (int s = 0; s < 8; s++) {
            int ks = s * 32;
            Frag a;
            const f16* ap = percBase + lm * 256 + ks + 8 * half;
            a.q[0] = *(const uint4*)ap;
            a.q[1] = *(const uint4*)(ap + 16);
            #pragma unroll
            for (int t = 0; t < 8; t++) {
                Frag bf;
                const f16* bp = sW12 + (t * 16 + lm) * 256 + ks + 8 * half;
                bf.q[0] = *(const uint4*)bp;
                bf.q[1] = *(const uint4*)(bp + 16);
                acc[t] = __builtin_amdgcn_wmma_f32_16x16x32_f16(false, a.v, false, bf.v,
                                                                (short)0, acc[t], false, false);
            }
        }
        // relu + bias -> LDS (layout transpose for layer-2 A fragments)
        f16* ub = sU + tw * 16 * 128;
        #pragma unroll
        for (int t = 0; t < 8; t++) {
            float bias = ub1[t * 16 + lm];
            #pragma unroll
            for (int r = 0; r < 8; r++) {
                int p2 = r + 8 * half;
                ub[p2 * 128 + t * 16 + lm] = (f16)fmaxf(acc[t][r] + bias, 0.f);
            }
        }
        // layer 2: 16x128 @ 128x16
        v8f c2 = zero;
        for (int s = 0; s < 4; s++) {
            Frag a, bf;
            const f16* ap = ub + lm * 128 + s * 32 + 8 * half;
            a.q[0] = *(const uint4*)ap;
            a.q[1] = *(const uint4*)(ap + 16);
            const f16* bp = sW12 + 32768 + lm * 128 + s * 32 + 8 * half;
            bf.q[0] = *(const uint4*)bp;
            bf.q[1] = *(const uint4*)(bp + 16);
            c2 = __builtin_amdgcn_wmma_f32_16x16x32_f16(false, a.v, false, bf.v,
                                                        (short)0, c2, false, false);
        }
        // x_new = x + dx * fire_mask
        int ch = lm;
        float bias2 = ub2[ch];
        int w0 = tw * 16;
        #pragma unroll
        for (int r = 0; r < 8; r++) {
            int px = w0 + r + 8 * half;
            uint32_t pidx = (uint32_t)((b * 64 + h) * 64 + px);
            float mask = (fireHash(pidx) <= 0.5f) ? 1.f : 0.f;
            int gi = (int)pidx * 16 + ch;
            xnew[gi] = x[gi] + (c2[r] + bias2) * mask;
        }
    }
}

// ---------------- k_final: living mask (pre & post 3x3 alpha max) -----------
__global__ void k_final(const float* __restrict__ x, const float* __restrict__ xnew,
                        float* __restrict__ out) {
    int idx = blockIdx.x * 256 + threadIdx.x;
    if (idx >= NB * 64 * 64) return;
    int w = idx & 63; int t = idx >> 6; int h = t & 63; int b = t >> 6;
    float preM = -1e30f, postM = -1e30f;
    for (int dy = -1; dy <= 1; dy++) {
        int y = h + dy; if ((unsigned)y >= 64u) continue;
        for (int dx = -1; dx <= 1; dx++) {
            int xx = w + dx; if ((unsigned)xx >= 64u) continue;
            int base = ((b * 64 + y) * 64 + xx) * 16 + 3;
            preM = fmaxf(preM, x[base]);
            postM = fmaxf(postM, xnew[base]);
        }
    }
    float living = (preM > 0.1f && postM > 0.1f) ? 1.f : 0.f;
    const float4* src = (const float4*)(xnew + idx * 16);
    float4* dst = (float4*)(out + idx * 16);
    #pragma unroll
    for (int c = 0; c < 4; c++) {
        float4 v = src[c];
        v.x *= living; v.y *= living; v.z *= living; v.w *= living;
        dst[c] = v;
    }
}

extern "C" void kernel_launch(void* const* d_in, const int* in_sizes, int n_in,
                              void* d_out, int out_size, void* d_ws, size_t ws_size,
                              hipStream_t stream) {
    const float* x      = (const float*)d_in[0];
    const float* images = (const float*)d_in[1];
    const float* bn1g = (const float*)d_in[2];
    const float* bn1b = (const float*)d_in[3];
    const float* bn1m = (const float*)d_in[4];
    const float* bn1v = (const float*)d_in[5];
    const float* c1w  = (const float*)d_in[6];
    const float* c1b  = (const float*)d_in[7];
    const float* bn2g = (const float*)d_in[8];
    const float* bn2b = (const float*)d_in[9];
    const float* bn2m = (const float*)d_in[10];
    const float* bn2v = (const float*)d_in[11];
    const float* c2w  = (const float*)d_in[12];
    const float* c2b  = (const float*)d_in[13];
    const float* dw   = (const float*)d_in[14];
    const float* db   = (const float*)d_in[15];
    const float* w1   = (const float*)d_in[16];
    const float* ub1  = (const float*)d_in[17];
    const float* w2   = (const float*)d_in[18];
    const float* ub2  = (const float*)d_in[19];

    char* ws = (char*)d_ws;
    float* h1     = (float*)(ws + WS_H1);
    float* xnew   = (float*)(ws + WS_XNEW);
    f16*   featA  = (f16*)(ws + WS_FEATA);
    f16*   filtbt = (f16*)(ws + WS_FILTBT);
    f16*   w1t    = (f16*)(ws + WS_W1T);
    f16*   w2t    = (f16*)(ws + WS_W2T);
    f16*   dwt    = (f16*)(ws + WS_DWT);
    float* out    = (float*)d_out;

    k_init<<<1280, 256, 0, stream>>>(w1, w2, w1t, w2t, filtbt);
    k_wtrans<<<(ND * FLAT + 255) / 256, 256, 0, stream>>>(dw, dwt);
    k_conv1<<<(NB * 36 * 36 * 32 + 127) / 128, 128, 0, stream>>>(
        images, bn1g, bn1b, bn1m, bn1v, c1w, c1b, h1);
    k_conv2<<<(NB * 18 * 18 * 32 + 127) / 128, 128, 0, stream>>>(
        h1, bn2g, bn2b, bn2m, bn2v, c2w, c2b, featA);
    k_dense<<<108, 128, 0, stream>>>(featA, dwt, db, filtbt);
    k_update<<<NB * 64, 128, 0, stream>>>(x, filtbt, w1t, ub1, ub2, xnew);
    k_final<<<(NB * 64 * 64 + 255) / 256, 256, 0, stream>>>(x, xnew, out);
}
